// GDANET_20710332301405
// MI455X (gfx1250) — matrix operand order repaired and checked by reading, not compiled, not run on hardware
//
#include <hip/hip_runtime.h>
#include <hip/hip_bf16.h>

typedef __attribute__((ext_vector_type(16))) _Float16 v16h;
typedef __attribute__((ext_vector_type(8)))  float    v8f;

#define B_   16
#define N_   1024
#define KL_  30
#define KG_  64
#define M_   256
#define BN_  (B_ * N_)
#define P1_  (B_ * N_ * KL_)

// ---------------------------------------------------------------------------
// WMMA tile index helpers (layouts this code uses, wave32; A/B k order is contiguous, not AMD's table order)
// A 16x32 f16: lane L -> M = L%16 ; element e -> K = 16g + e (one contiguous run, matches B)
// B 32x16 f16: lane L -> N = L%16 ; element e -> K = 16g + e (one contiguous run)
// D 16x16 f32: lane L -> N = L%16 ; element e -> M = e + 8g
// ---------------------------------------------------------------------------
__device__ __forceinline__ int a_kofs(int e, int g) {
  return 16 * g + e;
}

// ---------------------------------------------------------------------------
// Generic GEMM vs weight matrix: out[m,o] = sum_k A[m,k] * W[o,k] + bias[o]
// A: (Mtot x K) row-major (TA = float or _Float16), W: (O x K) f32 row-major.
// One wave per 16x16 tile. Branchless clamped loads; guarded tail chunk only.
// ---------------------------------------------------------------------------
template <typename TA, typename TO>
__global__ void gemm_w_kern(const TA* __restrict__ A, const float* __restrict__ W,
                            const float* __restrict__ bias, TO* __restrict__ out,
                            long long Mtot, int K, int O) {
  int gw   = (int)((blockIdx.x * (long long)blockDim.x + threadIdx.x) >> 5);
  int lane = threadIdx.x & 31;
  long long mt = (Mtot + 15) >> 4;
  int ot = (O + 15) >> 4;
  if ((long long)gw >= mt * ot) return;
  long long tm = gw % mt;
  int to = (int)(gw / mt);
  int g = lane >> 4, l = lane & 15;
  long long mrow = tm * 16 + l;
  int ocol = to * 16 + l;
  const TA*    Ar = A + (mrow < Mtot ? mrow : Mtot - 1) * (long long)K;
  const float* Wr = W + (long long)(ocol < O ? ocol : O - 1) * K;
  v8f acc = {};
  const int Kmain = K & ~31;
#pragma unroll 2
  for (int k0 = 0; k0 < Kmain; k0 += 32) {
    __builtin_prefetch((const void*)(Ar + k0 + 32), 0, 0);
    v16h a, b;
#pragma unroll
    for (int e = 0; e < 16; ++e) {
      a[e] = (_Float16)(float)Ar[k0 + a_kofs(e, g)];
      b[e] = (_Float16)Wr[k0 + 16 * g + e];
    }
    acc = __builtin_amdgcn_wmma_f32_16x16x32_f16(false, a, false, b, (short)0, acc,
                                                 false, false);
  }
  if (Kmain < K) {  // tail: clamp address, zero-select value (branchless)
    v16h a, b;
#pragma unroll
    for (int e = 0; e < 16; ++e) {
      int ka = Kmain + a_kofs(e, g);
      float av = (float)Ar[ka < K ? ka : K - 1];
      a[e] = (_Float16)(ka < K ? av : 0.f);
      int kb = Kmain + 16 * g + e;
      float bv = Wr[kb < K ? kb : K - 1];
      b[e] = (_Float16)(kb < K ? bv : 0.f);
    }
    acc = __builtin_amdgcn_wmma_f32_16x16x32_f16(false, a, false, b, (short)0, acc,
                                                 false, false);
  }
  if (ocol < O) {
    float bv = bias ? bias[ocol] : 0.f;
#pragma unroll
    for (int e = 0; e < 8; ++e) {
      long long m = tm * 16 + e + 8 * g;
      if (m < Mtot) out[m * O + ocol] = (TO)(acc[e] + bv);
    }
  }
}

// ---------------------------------------------------------------------------
// EdgeConv GEMM: A row m = (b,n,kk) position; channel c of the edge feature
// is built on the fly:   c < C : x[b,j,c]-x[b,n,c] ; c >= C : x[b,j,c-C]
// where j = idx[b,n,kk].  K = 2*C.  Output f16 (Mtot x O).  Branchless loads.
// ---------------------------------------------------------------------------
__global__ void gemm_edge_kern(const float* __restrict__ X, const int* __restrict__ idx,
                               const float* __restrict__ W, const float* __restrict__ bias,
                               _Float16* __restrict__ out, int C, int O) {
  const long long Mtot = P1_;
  const int K = 2 * C;
  int gw   = (int)((blockIdx.x * (long long)blockDim.x + threadIdx.x) >> 5);
  int lane = threadIdx.x & 31;
  long long mt = (Mtot + 15) >> 4;
  int ot = (O + 15) >> 4;
  if ((long long)gw >= mt * ot) return;
  long long tm = gw % mt;
  int to = (int)(gw / mt);
  int g = lane >> 4, l = lane & 15;
  long long mrow = tm * 16 + l;
  long long mc = (mrow < Mtot) ? mrow : Mtot - 1;
  int b  = (int)(mc / (N_ * KL_));
  int r  = (int)(mc - (long long)b * (N_ * KL_));
  int n  = r / KL_, kk = r - n * KL_;
  int j  = idx[((long long)b * N_ + n) * KL_ + kk];
  const float* xb = X + (long long)b * N_ * C;
  const float* xj = xb + (long long)j * C;
  const float* xn = xb + (long long)n * C;
  int ocol = to * 16 + l;
  const float* Wr = W + (long long)(ocol < O ? ocol : O - 1) * K;
  v8f acc = {};
  const int Kmain = K & ~31;
#pragma unroll 2
  for (int k0 = 0; k0 < Kmain; k0 += 32) {
    v16h a, bv;
#pragma unroll
    for (int e = 0; e < 16; ++e) {
      int ka = k0 + a_kofs(e, g);      // ka < Kmain <= 2C
      bool lo = ka < C;
      int cc = lo ? ka : ka - C;       // always in [0, C)
      float vj = xj[cc];
      float vn = xn[cc];
      a[e] = (_Float16)(lo ? (vj - vn) : vj);
      bv[e] = (_Float16)Wr[k0 + 16 * g + e];
    }
    acc = __builtin_amdgcn_wmma_f32_16x16x32_f16(false, a, false, bv, (short)0, acc,
                                                 false, false);
  }
  if (Kmain < K) {
    v16h a, bv;
#pragma unroll
    for (int e = 0; e < 16; ++e) {
      int ka = Kmain + a_kofs(e, g);
      int kaX = ka < K ? ka : K - 1;
      bool lo = kaX < C;
      int cc = lo ? kaX : kaX - C;
      float vj = xj[cc];
      float vn = xn[cc];
      float av = lo ? (vj - vn) : vj;
      a[e] = (_Float16)(ka < K ? av : 0.f);
      int kb = Kmain + 16 * g + e;
      float wv = Wr[kb < K ? kb : K - 1];
      bv[e] = (_Float16)(kb < K ? wv : 0.f);
    }
    acc = __builtin_amdgcn_wmma_f32_16x16x32_f16(false, a, false, bv, (short)0, acc,
                                                 false, false);
  }
  if (ocol < O) {
    float bs = bias ? bias[ocol] : 0.f;
#pragma unroll
    for (int e = 0; e < 8; ++e) {
      long long m = tm * 16 + e + 8 * g;
      if (m < Mtot) out[m * O + ocol] = (_Float16)(acc[e] + bs);
    }
  }
}

// ---------------------------------------------------------------------------
// Batched GEMM (per-batch A and B tensors, f32 in/out through f16 WMMA):
//   TRANSB=true : C[b,m,n] = scale * sum_k A[b,m,k] * B[b,n,k]   (B: b x Nb x K)
//   TRANSB=false: C[b,m,n] = scale * sum_k A[b,m,k] * B[b,k,n]   (B: b x K x Nb)
// Used for: kNN gram (A=B=X, transB), attention th*ph^T, att*g.
// ---------------------------------------------------------------------------
template <bool TRANSB>
__global__ void gemm_bat_kern(const float* __restrict__ A, const float* __restrict__ Bm,
                              float* __restrict__ Cm, int Ma, int Nb, int K, float scale) {
  int gw   = (int)((blockIdx.x * (long long)blockDim.x + threadIdx.x) >> 5);
  int lane = threadIdx.x & 31;
  int mt = (Ma + 15) >> 4, nt = (Nb + 15) >> 4;
  int tpb = mt * nt;
  if (gw >= tpb * B_) return;
  int b = gw / tpb, t = gw - b * tpb;
  int tm = t % mt, tn = t / mt;
  const float* Ab = A + (long long)b * Ma * K;
  const float* Bb = Bm + (long long)b * (long long)Nb * K;  // same size either way
  int g = lane >> 4, l = lane & 15;
  int mrow = tm * 16 + l, ncol = tn * 16 + l;
  int ncolc = ncol < Nb ? ncol : Nb - 1;
  const float* Ar = Ab + (long long)(mrow < Ma ? mrow : Ma - 1) * K;
  const float* Br = TRANSB ? (Bb + (long long)ncolc * K) : (Bb + ncolc);
  v8f acc = {};
  const int Kmain = K & ~31;
#pragma unroll 2
  for (int k0 = 0; k0 < Kmain; k0 += 32) {
    __builtin_prefetch((const void*)(Ar + k0 + 32), 0, 0);
    v16h a, bv;
#pragma unroll
    for (int e = 0; e < 16; ++e) {
      a[e] = (_Float16)Ar[k0 + a_kofs(e, g)];
      int kb = k0 + 16 * g + e;
      float v = TRANSB ? Br[kb] : Br[(long long)kb * Nb];
      bv[e] = (_Float16)v;
    }
    acc = __builtin_amdgcn_wmma_f32_16x16x32_f16(false, a, false, bv, (short)0, acc,
                                                 false, false);
  }
  if (Kmain < K) {
    v16h a, bv;
#pragma unroll
    for (int e = 0; e < 16; ++e) {
      int ka = Kmain + a_kofs(e, g);
      float av = Ar[ka < K ? ka : K - 1];
      a[e] = (_Float16)(ka < K ? av : 0.f);
      int kb = Kmain + 16 * g + e;
      int kbc = kb < K ? kb : K - 1;
      float v = TRANSB ? Br[kbc] : Br[(long long)kbc * Nb];
      bv[e] = (_Float16)(kb < K ? v : 0.f);
    }
    acc = __builtin_amdgcn_wmma_f32_16x16x32_f16(false, a, false, bv, (short)0, acc,
                                                 false, false);
  }
  if (ncol < Nb) {
#pragma unroll
    for (int e = 0; e < 8; ++e) {
      int m = tm * 16 + e + 8 * g;
      if (m < Ma)
        Cm[(long long)b * Ma * Nb + (long long)m * Nb + ncol] = acc[e] * scale;
    }
  }
}

// ---------------------------------------------------------------------------
// Row top-k from gram matrix. neg_d2(i,j) = 2 G[i,j] - G[i,i] - G[j,j].
// One wave per row; 1024 candidates live in 32 registers/lane; eliminated
// entries tracked with a per-lane bitmask (keeps array in VGPRs).
// ---------------------------------------------------------------------------
__global__ void topk_rows_kern(const float* __restrict__ G, int* __restrict__ idxOut,
                               int kOut) {
  int row  = (int)((blockIdx.x * (long long)blockDim.x + threadIdx.x) >> 5);
  int lane = threadIdx.x & 31;
  if (row >= B_ * N_) return;
  int b = row >> 10, i = row & 1023;
  const float* Gb = G + (long long)b * N_ * N_;
  const float* Gr = Gb + (long long)i * N_;
  float dii = Gr[i];
  float v[32];
#pragma unroll
  for (int t = 0; t < 32; ++t) {
    int j = t * 32 + lane;
    v[t] = 2.f * Gr[j] - dii - Gb[(long long)j * N_ + j];
  }
  unsigned mask = 0u;
  for (int it = 0; it < kOut; ++it) {
    float best = -3.4e38f;
    int bj = N_;
#pragma unroll
    for (int t = 0; t < 32; ++t) {
      if (!((mask >> t) & 1u)) {
        int j = t * 32 + lane;
        if (v[t] > best || (v[t] == best && j < bj)) { best = v[t]; bj = j; }
      }
    }
#pragma unroll
    for (int s = 16; s > 0; s >>= 1) {
      float ov = __shfl_xor(best, s, 32);
      int   oj = __shfl_xor(bj, s, 32);
      if (ov > best || (ov == best && oj < bj)) { best = ov; bj = oj; }
    }
    if (lane == 0) idxOut[(long long)row * kOut + it] = bj;
    if ((bj & 31) == lane) mask |= 1u << (bj >> 5);
  }
}

// top-256 (desc) and bottom-256 of energy per batch; one wave per batch.
__global__ void topk_energy_kern(const float* __restrict__ E, int* __restrict__ idxS,
                                 int* __restrict__ idxG) {
  int b    = (int)((blockIdx.x * (long long)blockDim.x + threadIdx.x) >> 5);
  int lane = threadIdx.x & 31;
  if (b >= B_) return;
  const float* Eb = E + b * N_;
  float v[32];
#pragma unroll
  for (int t = 0; t < 32; ++t) v[t] = Eb[t * 32 + lane];
  unsigned mask = 0u;
  for (int it = 0; it < M_; ++it) {
    float best = -3.4e38f;
    int bj = N_;
#pragma unroll
    for (int t = 0; t < 32; ++t)
      if (!((mask >> t) & 1u)) {
        int j = t * 32 + lane;
        if (v[t] > best || (v[t] == best && j < bj)) { best = v[t]; bj = j; }
      }
#pragma unroll
    for (int s = 16; s > 0; s >>= 1) {
      float ov = __shfl_xor(best, s, 32);
      int   oj = __shfl_xor(bj, s, 32);
      if (ov > best || (ov == best && oj < bj)) { best = ov; bj = oj; }
    }
    if (lane == 0) idxS[b * M_ + it] = bj;
    if ((bj & 31) == lane) mask |= 1u << (bj >> 5);
  }
  mask = 0u;
  for (int it = 0; it < M_; ++it) {
    float best = 3.4e38f;
    int bj = N_;
#pragma unroll
    for (int t = 0; t < 32; ++t)
      if (!((mask >> t) & 1u)) {
        int j = t * 32 + lane;
        if (v[t] < best || (v[t] == best && j < bj)) { best = v[t]; bj = j; }
      }
#pragma unroll
    for (int s = 16; s > 0; s >>= 1) {
      float ov = __shfl_xor(best, s, 32);
      int   oj = __shfl_xor(bj, s, 32);
      if (ov < best || (ov == best && oj < bj)) { best = ov; bj = oj; }
    }
    if (lane == 0) idxG[b * M_ + it] = bj;
    if ((bj & 31) == lane) mask |= 1u << (bj >> 5);
  }
}

// ---------------------------------------------------------------------------
// GDM high-pass energy: thresholded Gaussian adjacency over 64 neighbors in
// 64-dim feature space; energy = || sum_k A_k (x_j - x_n) ||^2.
// One thread per point; 64-channel accumulators fully unrolled in registers.
// ---------------------------------------------------------------------------
__global__ void gdm_energy_kern(const float* __restrict__ X, const int* __restrict__ idx,
                                float* __restrict__ E) {
  int p = blockIdx.x * blockDim.x + threadIdx.x;
  if (p >= B_ * N_) return;
  int b = p >> 10, n = p & 1023;
  const float* xb = X + (long long)b * N_ * 64;
  const float* xn = xb + (long long)n * 64;
  float xc[64], acc[64];
#pragma unroll
  for (int c = 0; c < 64; ++c) { xc[c] = xn[c]; acc[c] = 0.f; }
  float sumW = 0.f;
  const int* ir = idx + (long long)p * KG_;
  for (int kk = 0; kk < KG_; ++kk) {
    int j = ir[kk];
    const float* xj = xb + (long long)j * 64;
    float d2 = 1e-12f;
#pragma unroll
    for (int c = 0; c < 64; ++c) { float t = xj[c] - xc[c]; d2 += t * t; }
    float w = (d2 < 0.04f) ? __expf(-d2 * 0.25f) : 0.f;  // sigma^2 = 4, tau = 0.2
    sumW += w;
#pragma unroll
    for (int c = 0; c < 64; ++c) acc[c] += w * xj[c];
  }
  float inv = 1.f / (sumW + 1e-12f);
  float e = 0.f;
#pragma unroll
  for (int c = 0; c < 64; ++c) {
    float f = acc[c] * inv - sumW * inv * xc[c];
    e += f * f;
  }
  E[p] = e;
}

// ---------------------------------------------------------------------------
// BatchNorm statistics over positions (channel-last layout): one block/channel.
// ---------------------------------------------------------------------------
template <typename TA>
__global__ void bn_stats_kern(const TA* __restrict__ A, long long P, int C,
                              float* __restrict__ mean, float* __restrict__ rstd) {
  int c = blockIdx.x;
  __shared__ float s1[256], s2[256];
  float sum = 0.f, sq = 0.f;
  for (long long p = threadIdx.x; p < P; p += blockDim.x) {
    float v = (float)A[p * C + c];
    sum += v;
    sq += v * v;
  }
  s1[threadIdx.x] = sum;
  s2[threadIdx.x] = sq;
  __syncthreads();
  for (int s = 128; s > 0; s >>= 1) {
    if ((int)threadIdx.x < s) {
      s1[threadIdx.x] += s1[threadIdx.x + s];
      s2[threadIdx.x] += s2[threadIdx.x + s];
    }
    __syncthreads();
  }
  if (threadIdx.x == 0) {
    float m = s1[0] / (float)P;
    float var = s2[0] / (float)P - m * m;
    if (var < 0.f) var = 0.f;
    mean[c] = m;
    rstd[c] = rsqrtf(var + 1e-5f);
  }
}

template <typename TA, typename TO, bool RELU>
__global__ void bn_apply_kern(const TA* __restrict__ A, TO* __restrict__ Y,
                              const float* __restrict__ mean, const float* __restrict__ rstd,
                              const float* __restrict__ gamma, const float* __restrict__ beta,
                              const float* __restrict__ resid, long long total, int C) {
  long long i = blockIdx.x * (long long)blockDim.x + threadIdx.x;
  if (i >= total) return;
  int c = (int)(i % C);
  float v = (float)A[i];
  v = gamma[c] * (v - mean[c]) * rstd[c] + beta[c];
  if (RELU) v = fmaxf(v, 0.f);
  if (resid) v += resid[i];
  Y[i] = (TO)v;
}

// Fused BN + ReLU + max over the k (=30) neighbor axis: (B,N,k,C)f16 -> (B,N,C)f32.
__global__ void bn_relu_maxk_kern(const _Float16* __restrict__ A, float* __restrict__ Y,
                                  const float* __restrict__ mean, const float* __restrict__ rstd,
                                  const float* __restrict__ gamma, const float* __restrict__ beta,
                                  int C) {
  int i = blockIdx.x * blockDim.x + threadIdx.x;
  if (i >= BN_ * C) return;
  int c  = i % C;
  int pn = i / C;
  float sc = gamma[c] * rstd[c];
  float sh = beta[c] - gamma[c] * mean[c] * rstd[c];
  float m = 0.f;  // ReLU floor
  const _Float16* base = A + (long long)pn * KL_ * C + c;
  for (int kk = 0; kk < KL_; ++kk) {
    float v = sc * (float)base[(long long)kk * C] + sh;
    m = fmaxf(m, v);
  }
  Y[i] = m;
}

__global__ void gather_kern(const float* __restrict__ X, const int* __restrict__ idx,
                            float* __restrict__ Y, int C) {
  int i = blockIdx.x * blockDim.x + threadIdx.x;
  if (i >= B_ * M_ * C) return;
  int c = i % C, r = i / C;
  int b = r / M_, mi = r - b * M_;
  int j = idx[b * M_ + mi];
  Y[i] = X[((long long)b * N_ + j) * C + c];
}

__global__ void copych_kern(const float* __restrict__ S, float* __restrict__ D,
                            long long P, int Cs, int Cd, int off) {
  long long i = blockIdx.x * (long long)blockDim.x + threadIdx.x;
  if (i >= P * Cs) return;
  int c = (int)(i % Cs);
  long long p = i / Cs;
  D[p * Cd + off + c] = S[i];
}

// head: per (b,c) max and mean over N of feat (B,N,512) -> h (B,1024)
__global__ void maxmean_kern(const float* __restrict__ F, float* __restrict__ H) {
  int b = blockIdx.x >> 9;
  int c = blockIdx.x & 511;
  __shared__ float smax[256], ssum[256];
  float mx = -3.4e38f, sm = 0.f;
  for (int n = threadIdx.x; n < N_; n += 256) {
    float v = F[((long long)b * N_ + n) * 512 + c];
    mx = fmaxf(mx, v);
    sm += v;
  }
  smax[threadIdx.x] = mx;
  ssum[threadIdx.x] = sm;
  __syncthreads();
  for (int s = 128; s > 0; s >>= 1) {
    if ((int)threadIdx.x < s) {
      smax[threadIdx.x] = fmaxf(smax[threadIdx.x], smax[threadIdx.x + s]);
      ssum[threadIdx.x] += ssum[threadIdx.x + s];
    }
    __syncthreads();
  }
  if (threadIdx.x == 0) {
    H[b * 1024 + c] = smax[0];
    H[b * 1024 + 512 + c] = ssum[0] / (float)N_;
  }
}

// head BN over batch axis only (16 samples), + ReLU, in place.
__global__ void headbn_kern(float* __restrict__ A, const float* __restrict__ g,
                            const float* __restrict__ bt, int C) {
  int c = blockIdx.x * blockDim.x + threadIdx.x;
  if (c >= C) return;
  float s = 0.f, q = 0.f;
  for (int r = 0; r < B_; ++r) {
    float v = A[r * C + c];
    s += v;
    q += v * v;
  }
  float m = s / (float)B_;
  float var = q / (float)B_ - m * m;
  if (var < 0.f) var = 0.f;
  float rs = rsqrtf(var + 1e-5f);
  for (int r = 0; r < B_; ++r) {
    float v = g[c] * (A[r * C + c] - m) * rs + bt[c];
    A[r * C + c] = fmaxf(v, 0.f);
  }
}

// ---------------------------------------------------------------------------
// Host orchestration
// ---------------------------------------------------------------------------
enum {
  P_CONV1 = 2, P_CONV11 = 6, P_CONV12 = 10,
  P_CONV2 = 14, P_CONV21 = 18, P_CONV22 = 22,
  P_CONV3 = 26, P_CONV31 = 30, P_CONV32 = 34,
  P_CONV4 = 38,
  P_S1S = 42, P_S1G = 52, P_S2S = 62, P_S2G = 72,
  P_LIN1W = 82, P_LIN1B = 83, P_LIN2W = 84, P_LIN2B = 85,
  P_LIN3W = 86, P_LIN3B = 87,
  P_BN6G = 88, P_BN6B = 89, P_BN7G = 90, P_BN7B = 91
};

static inline unsigned nblk(long long total, int per) {
  return (unsigned)((total + per - 1) / per);
}
static inline unsigned gemm_blocks(long long Mtot, int O) {
  long long waves = ((Mtot + 15) / 16) * (long long)((O + 15) / 16);
  return (unsigned)((waves + 3) / 4);
}

extern "C" void kernel_launch(void* const* d_in, const int* in_sizes, int n_in,
                              void* d_out, int out_size, void* d_ws, size_t ws_size,
                              hipStream_t stream) {
  (void)in_sizes; (void)n_in; (void)out_size;
  const float* X0 = (const float*)d_in[0];  // (B,N,3)
  auto PF = [&](int i) { return (const float*)d_in[i]; };

  // ---- workspace bump allocator ----
  char* ws = (char*)d_ws;
  size_t off = 0;
  auto alloc = [&](size_t bytes) -> void* {
    void* p = ws + off;
    off += (bytes + 255) & ~(size_t)255;
    return p;
  };
  float*     G      = (float*)alloc((size_t)B_ * N_ * N_ * 4);
  _Float16*  bufA   = (_Float16*)alloc((size_t)P1_ * 128 * 2);
  _Float16*  bufB   = (_Float16*)alloc((size_t)P1_ * 128 * 2);
  int*       idx30  = (int*)alloc((size_t)BN_ * KL_ * 4);
  int*       idx64  = (int*)alloc((size_t)BN_ * KG_ * 4);
  float*     energy = (float*)alloc((size_t)BN_ * 4);
  int*       idxs   = (int*)alloc((size_t)B_ * M_ * 4);
  int*       idxg   = (int*)alloc((size_t)B_ * M_ * 4);
  float*     x1     = (float*)alloc((size_t)BN_ * 64 * 4);
  float*     x1t    = (float*)alloc((size_t)BN_ * 67 * 4);
  float*     x2     = (float*)alloc((size_t)BN_ * 64 * 4);
  float*     x2t    = (float*)alloc((size_t)BN_ * 131 * 4);
  float*     x3     = (float*)alloc((size_t)BN_ * 128 * 4);
  float*     z1     = (float*)alloc((size_t)BN_ * 64 * 4);
  float*     z2     = (float*)alloc((size_t)BN_ * 64 * 4);
  float*     z3     = (float*)alloc((size_t)BN_ * 128 * 4);
  float*     xs     = (float*)alloc((size_t)B_ * M_ * 64 * 4);
  float*     xg     = (float*)alloc((size_t)B_ * M_ * 64 * 4);
  float*     thb    = (float*)alloc((size_t)BN_ * 32 * 4);
  float*     phb    = (float*)alloc((size_t)B_ * M_ * 32 * 4);
  float*     gb     = (float*)alloc((size_t)B_ * M_ * 32 * 4);
  float*     yb     = (float*)alloc((size_t)BN_ * 32 * 4);
  float*     wy     = (float*)alloc((size_t)BN_ * 64 * 4);
  float*     ys     = (float*)alloc((size_t)BN_ * 64 * 4);
  float*     yg     = (float*)alloc((size_t)BN_ * 64 * 4);
  float*     cat128 = (float*)alloc((size_t)BN_ * 128 * 4);
  float*     cat256 = (float*)alloc((size_t)BN_ * 256 * 4);
  float*     featb  = (float*)alloc((size_t)BN_ * 512 * 4);
  float*     meanb  = (float*)alloc(512 * 4);
  float*     rstdb  = (float*)alloc(512 * 4);
  float*     h      = (float*)alloc((size_t)B_ * 1024 * 4);
  float*     h1     = (float*)alloc((size_t)B_ * 512 * 4);
  float*     h2     = (float*)alloc((size_t)B_ * 256 * 4);
  float*     att    = G;  // attention matrices reuse the (then-free) gram buffer
  if (off > ws_size) return;

  // ---- SGCAM: non-local attention with inter dim 32 ----
  auto sgcam = [&](int pb, const float* xf, const float* x2f, float* yout) {
    gemm_w_kern<float, float><<<gemm_blocks(B_ * M_, 32), 128, 0, stream>>>(
        x2f, PF(pb + 0), PF(pb + 1), gb, B_ * M_, 64, 32);
    gemm_w_kern<float, float><<<gemm_blocks(BN_, 32), 128, 0, stream>>>(
        xf, PF(pb + 2), PF(pb + 3), thb, BN_, 64, 32);
    gemm_w_kern<float, float><<<gemm_blocks(B_ * M_, 32), 128, 0, stream>>>(
        x2f, PF(pb + 4), PF(pb + 5), phb, B_ * M_, 64, 32);
    {  // att = th^T ph / M   (per-batch N x M)
      long long waves = (long long)B_ * (N_ / 16) * (M_ / 16);
      gemm_bat_kern<true><<<nblk(waves * 32, 128), 128, 0, stream>>>(
          thb, phb, att, N_, M_, 32, 1.f / (float)M_);
    }
    {  // y = att * g   (per-batch N x 32)
      long long waves = (long long)B_ * (N_ / 16) * (32 / 16);
      gemm_bat_kern<false><<<nblk(waves * 32, 128), 128, 0, stream>>>(
          att, gb, yb, N_, 32, M_, 1.f);
    }
    gemm_w_kern<float, float><<<gemm_blocks(BN_, 64), 128, 0, stream>>>(
        yb, PF(pb + 6), PF(pb + 7), wy, BN_, 32, 64);
    bn_stats_kern<float><<<dim3(64), dim3(256), 0, stream>>>(wy, BN_, 64, meanb, rstdb);
    bn_apply_kern<float, float, false><<<nblk((long long)BN_ * 64, 256), 256, 0, stream>>>(
        wy, yout, meanb, rstdb, PF(pb + 8), PF(pb + 9), xf, (long long)BN_ * 64, 64);
  };

  // ---- EdgeConv block: gram -> knn30 -> conv+BN+ReLU -> conv+BN+ReLU -> maxk ----
  auto edge_block = [&](const float* Xf, int C, int O, int pA, int pB, float* xout) {
    {  // gram(X X^T)
      long long waves = (long long)B_ * (N_ / 16) * (N_ / 16);
      gemm_bat_kern<true><<<nblk(waves * 32, 128), 128, 0, stream>>>(
          Xf, Xf, G, N_, N_, C, 1.f);
    }
    topk_rows_kern<<<nblk((long long)BN_ * 32, 128), 128, 0, stream>>>(G, idx30, KL_);
    gemm_edge_kern<<<gemm_blocks(P1_, O), 128, 0, stream>>>(
        Xf, idx30, PF(pA + 0), PF(pA + 1), bufA, C, O);
    bn_stats_kern<_Float16><<<dim3(O), dim3(256), 0, stream>>>(bufA, P1_, O, meanb, rstdb);
    bn_apply_kern<_Float16, _Float16, true><<<nblk((long long)P1_ * O, 256), 256, 0, stream>>>(
        bufA, bufA, meanb, rstdb, PF(pA + 2), PF(pA + 3), nullptr, (long long)P1_ * O, O);
    gemm_w_kern<_Float16, _Float16><<<gemm_blocks(P1_, O), 128, 0, stream>>>(
        bufA, PF(pB + 0), PF(pB + 1), bufB, P1_, O, O);
    bn_stats_kern<_Float16><<<dim3(O), dim3(256), 0, stream>>>(bufB, P1_, O, meanb, rstdb);
    bn_relu_maxk_kern<<<nblk((long long)BN_ * O, 256), 256, 0, stream>>>(
        bufB, xout, meanb, rstdb, PF(pB + 2), PF(pB + 3), O);
  };

  // ---- GDM + two SGCAMs + fuse conv (conv12/conv22) ----
  auto gdm_sgcam_fuse = [&](const float* xf, int pS, int pGm, int pFuse, float* zout) {
    {  // gram on 64-ch features + knn64
      long long waves = (long long)B_ * (N_ / 16) * (N_ / 16);
      gemm_bat_kern<true><<<nblk(waves * 32, 128), 128, 0, stream>>>(
          xf, xf, G, N_, N_, 64, 1.f);
    }
    topk_rows_kern<<<nblk((long long)BN_ * 32, 128), 128, 0, stream>>>(G, idx64, KG_);
    gdm_energy_kern<<<nblk(BN_, 256), 256, 0, stream>>>(xf, idx64, energy);
    topk_energy_kern<<<nblk((long long)B_ * 32, 128), 128, 0, stream>>>(energy, idxs, idxg);
    gather_kern<<<nblk((long long)B_ * M_ * 64, 256), 256, 0, stream>>>(xf, idxs, xs, 64);
    gather_kern<<<nblk((long long)B_ * M_ * 64, 256), 256, 0, stream>>>(xf, idxg, xg, 64);
    sgcam(pS, xf, xs, ys);
    sgcam(pGm, xf, xg, yg);
    copych_kern<<<nblk((long long)BN_ * 64, 256), 256, 0, stream>>>(ys, cat128, BN_, 64, 128, 0);
    copych_kern<<<nblk((long long)BN_ * 64, 256), 256, 0, stream>>>(yg, cat128, BN_, 64, 128, 64);
    gemm_w_kern<float, float><<<gemm_blocks(BN_, 64), 128, 0, stream>>>(
        cat128, PF(pFuse + 0), PF(pFuse + 1), zout, BN_, 128, 64);
    bn_stats_kern<float><<<dim3(64), dim3(256), 0, stream>>>(zout, BN_, 64, meanb, rstdb);
    bn_apply_kern<float, float, true><<<nblk((long long)BN_ * 64, 256), 256, 0, stream>>>(
        zout, zout, meanb, rstdb, PF(pFuse + 2), PF(pFuse + 3), nullptr,
        (long long)BN_ * 64, 64);
  };

  // ======================= block 1 =======================
  edge_block(X0, 3, 64, P_CONV1, P_CONV11, x1);
  gdm_sgcam_fuse(x1, P_S1S, P_S1G, P_CONV12, z1);
  copych_kern<<<nblk((long long)BN_ * 3, 256), 256, 0, stream>>>(X0, x1t, BN_, 3, 67, 0);
  copych_kern<<<nblk((long long)BN_ * 64, 256), 256, 0, stream>>>(z1, x1t, BN_, 64, 67, 3);

  // ======================= block 2 =======================
  edge_block(x1t, 67, 64, P_CONV2, P_CONV21, x2);
  gdm_sgcam_fuse(x2, P_S2S, P_S2G, P_CONV22, z2);
  copych_kern<<<nblk((long long)BN_ * 67, 256), 256, 0, stream>>>(x1t, x2t, BN_, 67, 131, 0);
  copych_kern<<<nblk((long long)BN_ * 64, 256), 256, 0, stream>>>(z2, x2t, BN_, 64, 131, 67);

  // ======================= block 3 =======================
  edge_block(x2t, 131, 128, P_CONV3, P_CONV31, x3);
  gemm_w_kern<float, float><<<gemm_blocks(BN_, 128), 128, 0, stream>>>(
      x3, PF(P_CONV32 + 0), PF(P_CONV32 + 1), z3, BN_, 128, 128);
  bn_stats_kern<float><<<dim3(128), dim3(256), 0, stream>>>(z3, BN_, 128, meanb, rstdb);
  bn_apply_kern<float, float, true><<<nblk((long long)BN_ * 128, 256), 256, 0, stream>>>(
      z3, z3, meanb, rstdb, PF(P_CONV32 + 2), PF(P_CONV32 + 3), nullptr,
      (long long)BN_ * 128, 128);

  // ================= feature fuse + head =================
  copych_kern<<<nblk((long long)BN_ * 64, 256), 256, 0, stream>>>(z1, cat256, BN_, 64, 256, 0);
  copych_kern<<<nblk((long long)BN_ * 64, 256), 256, 0, stream>>>(z2, cat256, BN_, 64, 256, 64);
  copych_kern<<<nblk((long long)BN_ * 128, 256), 256, 0, stream>>>(z3, cat256, BN_, 128, 256, 128);
  gemm_w_kern<float, float><<<gemm_blocks(BN_, 512), 128, 0, stream>>>(
      cat256, PF(P_CONV4 + 0), PF(P_CONV4 + 1), featb, BN_, 256, 512);
  bn_stats_kern<float><<<dim3(512), dim3(256), 0, stream>>>(featb, BN_, 512, meanb, rstdb);
  bn_apply_kern<float, float, true><<<nblk((long long)BN_ * 512, 256), 256, 0, stream>>>(
      featb, featb, meanb, rstdb, PF(P_CONV4 + 2), PF(P_CONV4 + 3), nullptr,
      (long long)BN_ * 512, 512);
  maxmean_kern<<<dim3(B_ * 512), dim3(256), 0, stream>>>(featb, h);

  gemm_w_kern<float, float><<<gemm_blocks(B_, 512), 128, 0, stream>>>(
      h, PF(P_LIN1W), PF(P_LIN1B), h1, B_, 1024, 512);
  headbn_kern<<<nblk(512, 256), 256, 0, stream>>>(h1, PF(P_BN6G), PF(P_BN6B), 512);
  gemm_w_kern<float, float><<<gemm_blocks(B_, 256), 128, 0, stream>>>(
      h1, PF(P_LIN2W), PF(P_LIN2B), h2, B_, 512, 256);
  headbn_kern<<<nblk(256, 256), 256, 0, stream>>>(h2, PF(P_BN7G), PF(P_BN7B), 256);
  gemm_w_kern<float, float><<<gemm_blocks(B_, 40), 128, 0, stream>>>(
      h2, PF(P_LIN3W), PF(P_LIN3B), (float*)d_out, B_, 256, 40);
}
